// S4LowRankBlock_53858889892289
// MI455X (gfx1250) — matrix operand order repaired and checked by reading, compiled, not run on hardware
//
#include <hip/hip_runtime.h>
#include <hip/hip_bf16.h>

#define DT      0.1f
#define I_DIM   128
#define H_DIM   64
#define BATCH   8
#define SEQ     4096
#define PI_F    3.14159265358979323846f

#define XSTRIDE 4144                 // ushorts per xs row: mult of 8 (16B), 8288B % 256 = 96
#define KREV_N  4128                 // 4096 reversed K + 32 zero guard (j < 0)

typedef __bf16 v16bf __attribute__((ext_vector_type(16)));
typedef float  v8f   __attribute__((ext_vector_type(8)));
typedef unsigned int u32x4 __attribute__((ext_vector_type(4)));
typedef u32x4 u32x4_u __attribute__((aligned(2)));   // lane-dependent (unaligned) LDS runs

union BF16Frag {
    v16bf v;
    unsigned short u[16];
    u32x4 q[2];
};

__device__ __forceinline__ unsigned short f2bf(float f) {
    unsigned int u = __float_as_uint(f);
    unsigned int r = u + 0x7FFFu + ((u >> 16) & 1u);   // round-to-nearest-even
    return (unsigned short)(r >> 16);
}

// ---------------------------------------------------------------------------
// Kernel A: fused Cauchy reduction -> K_hat[i][k] (complex, interleaved f32).
// Never materializes the [I, L, H] complex intermediate (8.6 GB in the ref).
// g_k = -i*(2/dt)*tan(pi*k/L) (purely imaginary), c_k = 1 - i*tan(pi*k/L).
// ---------------------------------------------------------------------------
__global__ void s4_compute_khat(const float* __restrict__ Bm,
                                const float* __restrict__ Ct,
                                const float* __restrict__ Lam,
                                const float* __restrict__ Pm,
                                const float* __restrict__ Qm,
                                float* __restrict__ khat) {
    int idx = blockIdx.x * blockDim.x + threadIdx.x;
    if (idx >= I_DIM * SEQ) return;
    int i = idx >> 12;           // / SEQ
    int k = idx & (SEQ - 1);

    float th = PI_F * (float)k / (float)SEQ;     // theta/2
    float tn = __tanf(th);
    float gi = -(2.0f / DT) * tn;                // imag(g); real(g) == 0
    float gi2 = gi * gi;

    const float* lp = Lam + i * H_DIM;
    const float* bp = Bm  + i * H_DIM;
    const float* cp = Ct  + i * H_DIM;
    const float* pp = Pm  + i * H_DIM;
    const float* qp = Qm  + i * H_DIM;

    float s00r = 0.f, s00i = 0.f, s01r = 0.f, s01i = 0.f;
    float s10r = 0.f, s10i = 0.f, s11r = 0.f, s11i = 0.f;
    #pragma unroll 4
    for (int h = 0; h < H_DIM; ++h) {
        float lam = lp[h];
        float rd  = __frcp_rn(lam * lam + gi2);  // 1 / |g - lam|^2
        float tr  = -lam * rd;                   // Re(1/(g - lam))
        float ti  = -gi  * rd;                   // Im(1/(g - lam))
        float b = bp[h], ct = cp[h], p = pp[h], q = qp[h];
        float w00 = ct * b, w01 = ct * p, w10 = q * b, w11 = q * p;
        s00r += w00 * tr;  s00i += w00 * ti;
        s01r += w01 * tr;  s01i += w01 * ti;
        s10r += w10 * tr;  s10i += w10 * ti;
        s11r += w11 * tr;  s11i += w11 * ti;
    }
    // K_hat = c * (k00 - k01*(1+k11)*k10), c = 1 + i*ci with ci = gi*dt/2
    float t1r = 1.0f + s11r, t1i = s11i;
    float t2r = s01r * t1r - s01i * t1i;
    float t2i = s01r * t1i + s01i * t1r;
    float t3r = t2r * s10r - t2i * s10i;
    float t3i = t2r * s10i + t2i * s10r;
    float nr = s00r - t3r, ni = s00i - t3i;
    float ci = gi * (DT * 0.5f);                 // = -tan(theta/2)
    khat[2 * idx + 0] = nr - ci * ni;
    khat[2 * idx + 1] = ci * nr + ni;
}

// ---------------------------------------------------------------------------
// Kernel B: irfft via direct real DFT over L/2+1 bins, twiddle recurrence
// resynced every 64 steps with exact integer-mod sincos.
// ---------------------------------------------------------------------------
__global__ void s4_irfft(const float* __restrict__ khat, float* __restrict__ Kout) {
    int idx = blockIdx.x * blockDim.x + threadIdx.x;
    if (idx >= I_DIM * SEQ) return;
    int i = idx >> 12;
    int n = idx & (SEQ - 1);

    const float* H = khat + (size_t)i * SEQ * 2;
    float acc = H[0];                                      // Re H_0
    float nyq = H[2 * (SEQ / 2)];                          // Re H_{L/2}
    acc += (n & 1) ? -nyq : nyq;

    const float step = 6.283185307179586476f / (float)SEQ; // 2*pi/L
    float cphi, sphi;
    __sincosf(step * (float)n, &sphi, &cphi);              // rotation by phi = 2*pi*n/L
    float cs = cphi, sn = sphi;                            // e^{i*1*phi}
    for (int k = 1; k < SEQ / 2; ++k) {
        if ((k & 63) == 0) {                               // resync: exact (k*n) mod L
            int r = (k * n) & (SEQ - 1);
            __sincosf(step * (float)r, &sn, &cs);
        }
        float hr = H[2 * k], hi = H[2 * k + 1];
        acc += 2.0f * (hr * cs - hi * sn);
        float c2 = cs * cphi - sn * sphi;                  // rotate by phi
        sn = sn * cphi + cs * sphi;
        cs = c2;
    }
    Kout[idx] = acc * (1.0f / (float)SEQ);
}

// ---------------------------------------------------------------------------
// Kernel C: causal depthwise convolution as per-channel (batch x Toeplitz)
// GEMM on v_wmma_f32_16x16x32_bf16.
//   A[m,kk] = x[b=m, t0+kk, i]      (batch-major LDS rows -> 2x ds_load_b128)
//   B[kk,n] = K[i, l0+n-t0-kk]      (reversed+guarded K  -> 2x 16B LDS loads)
//   D[m,n]  = y[b=m, l0+n, i]
// One workgroup per channel; whole per-channel x (bf16) + reversed K staged
// once in LDS (~74.5 KB of the 320 KB WGP pool). Two independent accumulators
// break the WMMA->WMMA RAW chain so each wave keeps 2 WMMAs in flight.
// ---------------------------------------------------------------------------
__device__ __forceinline__ void conv_step(const unsigned short* __restrict__ xrow,
                                          const unsigned short* __restrict__ krev,
                                          bool real_batch, int hi, int nl,
                                          int l0, int t0, v8f& acc) {
    BF16Frag fa, fb;
    // A (16x32 bf16): elem e -> kk=e+16hi
    if (real_batch) {
        fa.q[0] = *(const u32x4*)(xrow + t0 + 16 * hi);       // 16B aligned
        fa.q[1] = *(const u32x4*)(xrow + t0 + 8 + 16 * hi);   // 16B aligned
    } else {
        fa.q[0] = (u32x4)0;                                   // N-padding rows
        fa.q[1] = (u32x4)0;
    }
    // B (32x16 bf16): elem e -> kk=e+16hi; B[kk,n]=K[l0+n-t0-kk]
    // = krev[(4095 - l0 - nl + t0 + 16hi) + e]  (ascending contiguous)
    const int base = (SEQ - 1) - l0 - nl + t0 + 16 * hi;
    const u32x4_u* bp = (const u32x4_u*)(krev + base);
    fb.q[0] = bp[0];
    fb.q[1] = bp[1];
    acc = __builtin_amdgcn_wmma_f32_16x16x32_bf16(
        false, fa.v, false, fb.v, (short)0, acc, false, false);
}

__global__ void __launch_bounds__(512)
s4_conv_wmma(const float* __restrict__ x, const float* __restrict__ Kf,
             float* __restrict__ y) {
    extern __shared__ unsigned short smem[];
    unsigned short* xs   = smem;                    // [BATCH][XSTRIDE] bf16, batch-major
    unsigned short* krev = smem + BATCH * XSTRIDE;  // [KREV_N]: krev[c]=K[4095-c], c>4095 -> 0

    const int i   = blockIdx.x;                     // channel
    const int tid = threadIdx.x;
    const int nth = blockDim.x;

    // Stage x[:, :, i] -> bf16 LDS rows (zero-padded tails), 2 elems per store
    for (int idx = tid; idx < BATCH * (XSTRIDE / 2); idx += nth) {
        int b = idx / (XSTRIDE / 2), t2 = idx - b * (XSTRIDE / 2);
        int t = 2 * t2;
        const float* xp = x + ((size_t)b * SEQ + t) * I_DIM + i;
        unsigned int lo = (t     < SEQ) ? f2bf(xp[0])     : 0u;
        unsigned int hiw = (t + 1 < SEQ) ? f2bf(xp[I_DIM]) : 0u;
        *(unsigned int*)(xs + b * XSTRIDE + t) = lo | (hiw << 16);
    }
    // Stage K[i, :] reversed with zero guard (covers j < 0 => causality)
    for (int c = tid; c < KREV_N; c += nth) {
        krev[c] = (c < SEQ) ? f2bf(Kf[(size_t)i * SEQ + (SEQ - 1 - c)])
                            : (unsigned short)0;
    }
    __syncthreads();

    const int lane = tid & 31;
    const int wave = tid >> 5;
    const int nw   = nth >> 5;
    const int nl   = lane & 15;      // A: row m (batch) ; B/D: col n (l offset)
    const int hi   = lane >> 4;      // half-wave selector per ISA layouts

    const unsigned short* xrow = xs + nl * XSTRIDE;
    const bool real_batch = (nl < BATCH);

    for (int tile = wave; tile < SEQ / 16; tile += nw) {
        const int l0 = tile * 16;
        v8f acc0 = {}, acc1 = {};
        int t0 = 0;
        // Dual-accumulator unroll: two independent WMMA chains in flight
        for (; t0 + 32 <= l0; t0 += 64) {
            conv_step(xrow, krev, real_batch, hi, nl, l0, t0,      acc0);
            conv_step(xrow, krev, real_batch, hi, nl, l0, t0 + 32, acc1);
        }
        for (; t0 <= l0; t0 += 32) {
            conv_step(xrow, krev, real_batch, hi, nl, l0, t0, acc0);
        }
        acc0 = acc0 + acc1;
        // D (16x16 f32): VGPR r -> M(batch) = r + 8*hi ; N = nl -> l = l0+nl
        if (hi == 0) {                       // M = 0..7 are the real batches
            #pragma unroll
            for (int r = 0; r < 8; ++r) {
                y[((size_t)r * SEQ + (l0 + nl)) * I_DIM + i] = acc0[r];
            }
        }
    }
}

extern "C" void kernel_launch(void* const* d_in, const int* in_sizes, int n_in,
                              void* d_out, int out_size, void* d_ws, size_t ws_size,
                              hipStream_t stream) {
    const float* x   = (const float*)d_in[0];
    const float* Bm  = (const float*)d_in[1];
    const float* Ct  = (const float*)d_in[2];
    const float* Lam = (const float*)d_in[3];   // [I,1,H] == [I,H] flat
    const float* Pm  = (const float*)d_in[4];
    const float* Qm  = (const float*)d_in[5];
    float* y = (float*)d_out;

    float* khat = (float*)d_ws;                         // [I][SEQ][2] f32 = 4 MB
    float* Kt   = khat + (size_t)I_DIM * SEQ * 2;       // [I][SEQ]    f32 = 2 MB

    const int total = I_DIM * SEQ;
    s4_compute_khat<<<total / 256, 256, 0, stream>>>(Bm, Ct, Lam, Pm, Qm, khat);
    s4_irfft<<<total / 256, 256, 0, stream>>>(khat, Kt);

    const size_t shmem = ((size_t)BATCH * XSTRIDE + KREV_N) * sizeof(unsigned short);
    s4_conv_wmma<<<I_DIM, 512, shmem, stream>>>(x, Kt, y);
}
